// AGCN_Block_74972949119136
// MI455X (gfx1250) — compile-verified, matrix operand order checked by reading, not run on hardware
//
#include <hip/hip_runtime.h>
#include <hip/hip_bf16.h>
#include <stdint.h>

// AGCN block for gfx1250: bf16 WMMA (f32 accum) for the two big GEMMs
// (stacked 1x1 conv K=192, temporal 9-tap conv K=1152 + fused residual GEMM),
// VALU for the small V=25-shaped graph/softmax stages.
// GEMM waves are register-blocked 16(M) x 64(N): each A (weight) fragment is
// reused across 4 column tiles -> 2.5 b128 loads per WMMA instead of 4.

typedef __attribute__((ext_vector_type(16))) __bf16 v16bf;
typedef __attribute__((ext_vector_type(8)))  float  v8f;

#define BN_EPS 1e-5f

__device__ __forceinline__ unsigned short f2bf(float f) {
  unsigned u = __float_as_uint(f);
  unsigned r = u + 0x7FFFu + ((u >> 16) & 1u);   // round-to-nearest-even
  return (unsigned short)(r >> 16);
}

union FragAB {                 // wave-fragment of a 16x32 (A) / 32x16 (B) bf16 tile
  v16bf v;
  uint4 q[2];
  unsigned short h[16];
};
union Out8 {
  uint4 q;
  unsigned short h[8];
};

// ---------------------------------------------------------------- weight prep
// W2eff[o,k] = sum_s' Ww[s'*128+o, k]  (folds the reshape+sum over subsets)
// WbEff[o]   = sum_s Wb[s*128+o]
// At[(o*9+kt)*128+c] = bf16(wt[o,c,kt,0])   (k-contiguous for A fragments)
// wrbf[o*64+c]       = bf16(wr[o,c])
__global__ void k_prep(const float* __restrict__ Ww, const float* __restrict__ Wb,
                       const float* __restrict__ wt, const float* __restrict__ wr,
                       unsigned short* __restrict__ W2bf, float* __restrict__ WbEff,
                       unsigned short* __restrict__ At, unsigned short* __restrict__ wrbf) {
  int stride = gridDim.x * blockDim.x;
  int tid0 = blockIdx.x * blockDim.x + threadIdx.x;
  for (int idx = tid0; idx < 128 * 192; idx += stride) {
    int o = idx / 192, k = idx % 192;
    float s = Ww[(size_t)(0 * 128 + o) * 192 + k]
            + Ww[(size_t)(1 * 128 + o) * 192 + k]
            + Ww[(size_t)(2 * 128 + o) * 192 + k];
    W2bf[idx] = f2bf(s);
  }
  for (int o = tid0; o < 128; o += stride)
    WbEff[o] = Wb[o] + Wb[128 + o] + Wb[256 + o];
  for (int idx = tid0; idx < 128 * 9 * 128; idx += stride) {
    int o = idx / 1152, r = idx % 1152, kt = r / 128, c = r % 128;
    At[idx] = f2bf(wt[(size_t)(o * 128 + c) * 9 + kt]);
  }
  for (int idx = tid0; idx < 128 * 64; idx += stride)
    wrbf[idx] = f2bf(wr[idx]);
}

// ---------------------------------------------------------------- x mean over T
__global__ void k_xmean(const float* __restrict__ x, float* __restrict__ xm) {
  int idx = blockIdx.x * blockDim.x + threadIdx.x;
  if (idx >= 64 * 64 * 25) return;
  int v = idx % 25, c = (idx / 25) % 64, n = idx / 1600;
  const float* p = x + (((size_t)n * 64 + c) * 256) * 25 + v;
  float s = 0.f;
  for (int t = 0; t < 256; ++t) s += p[(size_t)t * 25];
  xm[idx] = s * (1.0f / 256.0f);
}

// ---------------------------------------------------------------- adaptive adjacency
// th/ph GEMMs (128x64x25), Cmat = softmax(th^T ph), adp = A + B + Cmat. One block per n.
__global__ void k_adp(const float* __restrict__ xm,
                      const float* __restrict__ thw, const float* __restrict__ thb,
                      const float* __restrict__ phw, const float* __restrict__ phb,
                      const float* __restrict__ Ag, const float* __restrict__ Bg,
                      float* __restrict__ adp) {
  __shared__ float sxm[64 * 25];
  __shared__ float sth[128 * 25];
  __shared__ float sph[128 * 25];
  __shared__ float sC[625];
  int n = blockIdx.x, tid = threadIdx.x;
  for (int i = tid; i < 1600; i += 256) sxm[i] = xm[(size_t)n * 1600 + i];
  __syncthreads();
  for (int i = tid; i < 3200; i += 256) {
    int e = i / 25, v = i % 25;
    float a = thb[e], b = phb[e];
    for (int c = 0; c < 64; ++c) {
      float xv = sxm[c * 25 + v];
      a += thw[e * 64 + c] * xv;
      b += phw[e * 64 + c] * xv;
    }
    sth[i] = a; sph[i] = b;
  }
  __syncthreads();
  for (int i = tid; i < 625; i += 256) {
    int v = i / 25, w = i % 25;
    float acc = 0.f;
    for (int e = 0; e < 128; ++e) acc += sth[e * 25 + v] * sph[e * 25 + w];
    sC[i] = acc;
  }
  __syncthreads();
  if (tid < 25) {                       // row-wise softmax over w
    int v = tid;
    float mx = -3.4e38f;
    for (int w = 0; w < 25; ++w) mx = fmaxf(mx, sC[v * 25 + w]);
    float sum = 0.f;
    for (int w = 0; w < 25; ++w) { float e = __expf(sC[v * 25 + w] - mx); sC[v * 25 + w] = e; sum += e; }
    float inv = 1.0f / sum;
    for (int w = 0; w < 25; ++w) sC[v * 25 + w] *= inv;
  }
  __syncthreads();
  for (int i = tid; i < 1875; i += 256)
    adp[(size_t)n * 1875 + i] = Ag[i] + Bg[i] + sC[i % 625];
}

// ---------------------------------------------------------------- graph aggregation
// aggT[n,(t,w), s*64+c] = bf16( sum_v x[n,c,t,v] * adp[n,s,v,w] )  (K-contiguous)
// also emits x_t[n,(t,v), c] = bf16(x) for the fused residual GEMM. One block per (n,t).
__global__ void k_agg(const float* __restrict__ x, const float* __restrict__ adp,
                      unsigned short* __restrict__ aggT, unsigned short* __restrict__ x_t) {
  __shared__ float sx[64 * 25];
  __shared__ float sa[3 * 25 * 25];
  int nt = blockIdx.x;
  int n = nt / 256, t = nt % 256, tid = threadIdx.x;
  for (int i = tid; i < 1600; i += 256) {
    int c = i / 25, v = i % 25;
    float val = x[(((size_t)n * 64 + c) * 256 + t) * 25 + v];
    sx[i] = val;
    x_t[(((size_t)n * 256 + t) * 25 + v) * 64 + c] = f2bf(val);
  }
  for (int i = tid; i < 1875; i += 256) sa[i] = adp[(size_t)n * 1875 + i];
  __syncthreads();
  for (int idx = tid; idx < 4800; idx += 256) {
    int c = idx & 63;
    int s = (idx >> 6) % 3;
    int w = idx / 192;
    float acc = 0.f;
    const float* xr = &sx[c * 25];
    const float* ar = &sa[s * 625 + w];
    for (int v = 0; v < 25; ++v) acc += xr[v] * ar[v * 25];
    aggT[(((size_t)n * 256 + t) * 25 + w) * 192 + s * 64 + c] = f2bf(acc);
  }
}

// ---------------------------------------------------------------- GEMM1 (stacked 1x1 conv)
// y_t[n,(t,v),o] = bf16(relu(BN_s( W2eff @ aggT + WbEff )))   M=128, K=192, 6400 cols/n.
// One wave per 16x64 block of outputs: 6 k-steps, A fragment reused over 4 col tiles.
__global__ void k_gemm1(const unsigned short* __restrict__ aggT,
                        const unsigned short* __restrict__ W2bf,
                        const float* __restrict__ WbEff,
                        const float* __restrict__ g_s, const float* __restrict__ b_s,
                        unsigned short* __restrict__ y_t) {
  const float inv = rsqrtf(1.0f + BN_EPS);
  int wave = (blockIdx.x * blockDim.x + threadIdx.x) >> 5;
  int lane = threadIdx.x & 31;
  int lo = lane & 15, hi = lane >> 4;
  int n = wave / 800;          // 8 oTiles * 100 col-groups per n
  int rem = wave % 800;
  int oT = rem / 100;
  int g  = rem % 100;          // 64-column group
  v8f acc[4] = {};
  const unsigned short* arow = W2bf + (size_t)(oT * 16 + lo) * 192 + 8 * hi;
  const unsigned short* brow[4];
#pragma unroll
  for (int j = 0; j < 4; ++j)
    brow[j] = aggT + ((size_t)n * 6400 + g * 64 + j * 16 + lo) * 192 + 8 * hi;
#pragma unroll
  for (int ks = 0; ks < 6; ++ks) {
    FragAB a;
    a.q[0] = *(const uint4*)(arow + ks * 32);
    a.q[1] = *(const uint4*)(arow + ks * 32 + 16);
    if (ks < 5) __builtin_prefetch(brow[0] + (ks + 1) * 32, 0, 3);
#pragma unroll
    for (int j = 0; j < 4; ++j) {
      FragAB b;
      b.q[0] = *(const uint4*)(brow[j] + ks * 32);
      b.q[1] = *(const uint4*)(brow[j] + ks * 32 + 16);
      acc[j] = __builtin_amdgcn_wmma_f32_16x16x32_bf16(false, a.v, false, b.v,
                                                       (short)0, acc[j], false, false);
    }
  }
  int o0 = oT * 16 + 8 * hi;              // lane covers 8 consecutive output channels
  float bia[8], scl[8], sft[8];
#pragma unroll
  for (int r = 0; r < 8; ++r) {
    int o = o0 + r;
    bia[r] = WbEff[o];
    scl[r] = g_s[o] * inv;
    sft[r] = b_s[o];
  }
#pragma unroll
  for (int j = 0; j < 4; ++j) {
    Out8 o8;
#pragma unroll
    for (int r = 0; r < 8; ++r) {
      float val = (acc[j][r] + bia[r]) * scl[r] + sft[r];
      o8.h[r] = f2bf(fmaxf(val, 0.f));
    }
    *(uint4*)(y_t + ((size_t)n * 6400 + g * 64 + j * 16 + lo) * 128 + o0) = o8.q;
  }
}

// ---------------------------------------------------------------- GEMM2 (temporal conv + fused residual)
// out = relu( BN_t( conv9x1(y) + bt ) + BN_r( wr@x + br ) )
// Wave computes 16(o) x 64(cols): main path 9 taps x 4 k-chunks x 4 col tiles = 144
// WMMAs with each A fragment reused 4x; residual adds 8 WMMAs on 2nd accumulator set.
__global__ void k_gemm2(const unsigned short* __restrict__ y_t,
                        const unsigned short* __restrict__ x_t,
                        const unsigned short* __restrict__ At,
                        const unsigned short* __restrict__ wrbf,
                        const float* __restrict__ bt,
                        const float* __restrict__ g_t, const float* __restrict__ b_t,
                        const float* __restrict__ br,
                        const float* __restrict__ g_r, const float* __restrict__ b_r,
                        float* __restrict__ out) {
  const float inv = rsqrtf(1.0f + BN_EPS);
  int wave = (blockIdx.x * blockDim.x + threadIdx.x) >> 5;
  int lane = threadIdx.x & 31;
  int lo = lane & 15, hi = lane >> 4;
  int n = wave / 800;
  int rem = wave % 800;
  int oT = rem / 100;
  int g  = rem % 100;
  int col[4], t[4], v[4];
#pragma unroll
  for (int j = 0; j < 4; ++j) {
    col[j] = g * 64 + j * 16 + lo;
    t[j] = col[j] / 25;
    v[j] = col[j] % 25;
  }
  v8f acc[4] = {}, acc2[4] = {};
  const size_t nbase = (size_t)n * 6400;
  const unsigned short* arowA = At + (size_t)(oT * 16 + lo) * 1152 + 8 * hi;
  for (int tap = 0; tap < 9; ++tap) {
    const unsigned short* brow[4];
    bool inb[4];
#pragma unroll
    for (int j = 0; j < 4; ++j) {
      int tt = t[j] + tap - 4;
      inb[j] = (tt >= 0) && (tt < 256);
      brow[j] = y_t + (nbase + (size_t)tt * 25 + v[j]) * 128 + 8 * hi;
    }
#pragma unroll
    for (int kc = 0; kc < 4; ++kc) {
      FragAB a;
      a.q[0] = *(const uint4*)(arowA + tap * 128 + kc * 32);
      a.q[1] = *(const uint4*)(arowA + tap * 128 + kc * 32 + 16);
#pragma unroll
      for (int j = 0; j < 4; ++j) {
        FragAB b;
        if (inb[j]) {
          b.q[0] = *(const uint4*)(brow[j] + kc * 32);
          b.q[1] = *(const uint4*)(brow[j] + kc * 32 + 16);
        } else {
          b.q[0] = make_uint4(0u, 0u, 0u, 0u);
          b.q[1] = make_uint4(0u, 0u, 0u, 0u);
        }
        acc[j] = __builtin_amdgcn_wmma_f32_16x16x32_bf16(false, a.v, false, b.v,
                                                         (short)0, acc[j], false, false);
      }
    }
  }
  const unsigned short* arowR = wrbf + (size_t)(oT * 16 + lo) * 64 + 8 * hi;
#pragma unroll
  for (int kc = 0; kc < 2; ++kc) {
    FragAB a;
    a.q[0] = *(const uint4*)(arowR + kc * 32);
    a.q[1] = *(const uint4*)(arowR + kc * 32 + 16);
#pragma unroll
    for (int j = 0; j < 4; ++j) {
      FragAB b;
      const unsigned short* browR = x_t + (nbase + col[j]) * 64 + 8 * hi;
      b.q[0] = *(const uint4*)(browR + kc * 32);
      b.q[1] = *(const uint4*)(browR + kc * 32 + 16);
      acc2[j] = __builtin_amdgcn_wmma_f32_16x16x32_bf16(false, a.v, false, b.v,
                                                        (short)0, acc2[j], false, false);
    }
  }
  int o0 = oT * 16 + 8 * hi;
  float btv[8], sT[8], bT[8], brv[8], sR[8], bR[8];
#pragma unroll
  for (int r = 0; r < 8; ++r) {
    int o = o0 + r;
    btv[r] = bt[o];  sT[r] = g_t[o] * inv;  bT[r] = b_t[o];
    brv[r] = br[o];  sR[r] = g_r[o] * inv;  bR[r] = b_r[o];
  }
#pragma unroll
  for (int j = 0; j < 4; ++j) {
    float* orow = out + ((size_t)n * 128 + o0) * 6400 + (size_t)t[j] * 25 + v[j];
#pragma unroll
    for (int r = 0; r < 8; ++r) {
      float m  = (acc[j][r]  + btv[r]) * sT[r] + bT[r];
      float rs = (acc2[j][r] + brv[r]) * sR[r] + bR[r];
      orow[(size_t)r * 6400] = fmaxf(m + rs, 0.f);
    }
  }
}

// ---------------------------------------------------------------- launcher
extern "C" void kernel_launch(void* const* d_in, const int* in_sizes, int n_in,
                              void* d_out, int out_size, void* d_ws, size_t ws_size,
                              hipStream_t stream) {
  const float* x   = (const float*)d_in[0];
  const float* Ag  = (const float*)d_in[1];
  const float* thw = (const float*)d_in[2];
  const float* thb = (const float*)d_in[3];
  const float* phw = (const float*)d_in[4];
  const float* phb = (const float*)d_in[5];
  const float* Bg  = (const float*)d_in[6];
  const float* Ww  = (const float*)d_in[7];
  const float* Wb  = (const float*)d_in[8];
  const float* g_s = (const float*)d_in[9];
  const float* b_s = (const float*)d_in[10];
  const float* wt  = (const float*)d_in[11];
  const float* bt  = (const float*)d_in[12];
  const float* g_t = (const float*)d_in[13];
  const float* b_t = (const float*)d_in[14];
  const float* wr  = (const float*)d_in[15];
  const float* br  = (const float*)d_in[16];
  const float* g_r = (const float*)d_in[17];
  const float* b_r = (const float*)d_in[18];
  (void)in_sizes; (void)n_in; (void)out_size; (void)ws_size;

  char* ws = (char*)d_ws;
  size_t off = 0;
  auto carve = [&](size_t bytes) -> char* {
    char* p = ws + off;
    off = (off + bytes + 255) & ~(size_t)255;
    return p;
  };
  float*          xm    = (float*)carve((size_t)64 * 64 * 25 * 4);
  float*          adp   = (float*)carve((size_t)64 * 3 * 25 * 25 * 4);
  unsigned short* W2bf  = (unsigned short*)carve((size_t)128 * 192 * 2);
  float*          WbEff = (float*)carve((size_t)128 * 4);
  unsigned short* At    = (unsigned short*)carve((size_t)128 * 9 * 128 * 2);
  unsigned short* wrbf  = (unsigned short*)carve((size_t)128 * 64 * 2);
  unsigned short* x_t   = (unsigned short*)carve((size_t)64 * 256 * 25 * 64 * 2);
  unsigned short* aggT  = (unsigned short*)carve((size_t)64 * 256 * 25 * 192 * 2);
  unsigned short* y_t   = (unsigned short*)carve((size_t)64 * 256 * 25 * 128 * 2);

  k_prep <<<72, 256, 0, stream>>>(Ww, Wb, wt, wr, W2bf, WbEff, At, wrbf);
  k_xmean<<<(64 * 64 * 25 + 255) / 256, 256, 0, stream>>>(x, xm);
  k_adp  <<<64, 256, 0, stream>>>(xm, thw, thb, phw, phb, Ag, Bg, adp);
  k_agg  <<<64 * 256, 256, 0, stream>>>(x, adp, aggT, x_t);
  // 64 n * 8 oTiles * 100 col-groups = 51200 waves, 8 waves/block
  k_gemm1<<<6400, 256, 0, stream>>>(aggT, W2bf, WbEff, g_s, b_s, y_t);
  k_gemm2<<<6400, 256, 0, stream>>>(y_t, x_t, At, wrbf, bt, g_t, b_t,
                                    br, g_r, b_r, (float*)d_out);
}
